// Decoder_78262894068043
// MI455X (gfx1250) — matrix operand order repaired and checked by reading, not compiled, not run on hardware
//
#include <hip/hip_runtime.h>
#include <stdint.h>

// ---------------- problem constants (match reference) ----------------
constexpr int Bc  = 256;     // batch
constexpr int Mm  = 32;      // measures
constexpr int Sc  = 16;      // steps per measure
constexpr int Tt  = Mm * Sc; // 512 beat steps
constexpr int INc = 512;     // input dim
constexpr int Hc  = 1024;    // hidden dim
constexpr int Oc  = 128;     // output dim
constexpr int G4H = 4 * Hc;  // 4096 gate width

// ---------------- types ----------------
typedef __bf16 bf16_t;
typedef __attribute__((ext_vector_type(16))) __bf16 v16bf;
typedef __attribute__((ext_vector_type(8)))  __bf16 v8bf;
typedef __attribute__((ext_vector_type(8)))  float  v8f;

__device__ __forceinline__ bf16_t f2bf(float x) {
    union { float f; uint32_t u; } v; v.f = x;
    uint32_t r = v.u + 0x7FFFu + ((v.u >> 16) & 1u);   // round-to-nearest-even
    union { uint16_t u; bf16_t b; } o; o.u = (uint16_t)(r >> 16);
    return o.b;
}

__device__ __forceinline__ float sigmoidf(float x) {
    return 1.0f / (1.0f + __expf(-x));
}

// ---------------- utility kernels ----------------
__global__ void cvt_f32_bf16_kernel(const float* __restrict__ in,
                                    bf16_t* __restrict__ out, long n) {
    long i = (long)blockIdx.x * blockDim.x + threadIdx.x;
    long stride = (long)gridDim.x * blockDim.x;
    for (; i < n; i += stride) out[i] = f2bf(in[i]);
}

__global__ void fill_zero_kernel(uint32_t* __restrict__ p, long n) {
    long i = (long)blockIdx.x * blockDim.x + threadIdx.x;
    long stride = (long)gridDim.x * blockDim.x;
    for (; i < n; i += stride) p[i] = 0u;
}

// ---------------- WMMA bf16 GEMM ----------------
// C[Mrows, Ncols] = op( A[Mrows,K] @ W[Ncols,K]^T  (+ A2[Mrows,K2] @ W2[Ncols,K2]^T) )
//                   + bias[col] (optional) + C (if accumulate)
// Per ISA: A frag 16x32, lane l=row, hi=lane>>4 selects K sub-blocks.
//          B frag 32x16 from W rows (W stored [N,K] row-major => B[k][n]=W[n][k]).
// Block = 256 threads = 8 waves (4 along M x 2 along N), wave tile 32x32,
// block tile 128(M) x 64(N). Mrows%128==0, Ncols%64==0, K%32==0 assumed.
__device__ __forceinline__ v16bf load_a_frag(const bf16_t* __restrict__ rowPtr, int hi) {
    union { v16bf v; v8bf h[2]; } u;
    u.h[0] = *reinterpret_cast<const v8bf*>(rowPtr + hi * 8);
    u.h[1] = *reinterpret_cast<const v8bf*>(rowPtr + 16 + hi * 8);
    return u.v;
}
__device__ __forceinline__ v16bf load_b_frag(const bf16_t* __restrict__ rowPtr, int hi) {
    union { v16bf v; v8bf h[2]; } u;
    u.h[0] = *reinterpret_cast<const v8bf*>(rowPtr + hi * 8);
    u.h[1] = *reinterpret_cast<const v8bf*>(rowPtr + 16 + hi * 8);
    return u.v;
}

__global__ __launch_bounds__(256)
void gemm_bf16_wmma(const bf16_t* __restrict__ A,  int lda,
                    const bf16_t* __restrict__ W,  int ldw,  int K,
                    const bf16_t* __restrict__ A2, int lda2,
                    const bf16_t* __restrict__ W2, int ldw2, int K2,
                    float* __restrict__ C, int ldc,
                    const float* __restrict__ bias, int accumulate)
{
    const int tid  = threadIdx.x;
    const int wave = tid >> 5;       // 0..7
    const int lane = tid & 31;
    const int l    = lane & 15;
    const int hi   = lane >> 4;

    const int wm = wave & 3;         // 0..3 along M
    const int wn = wave >> 2;        // 0..1 along N

    const int rowBase = blockIdx.y * 128 + wm * 32;  // two 16-row tiles
    const int colBase = blockIdx.x * 64  + wn * 32;  // two 16-col tiles

    v8f acc[2][2];
#pragma unroll
    for (int mi = 0; mi < 2; ++mi)
#pragma unroll
        for (int ni = 0; ni < 2; ++ni)
#pragma unroll
            for (int r = 0; r < 8; ++r) acc[mi][ni][r] = 0.0f;

    auto do_segment = [&](const bf16_t* __restrict__ Ab, int ldab,
                          const bf16_t* __restrict__ Wb, int ldwb, int Kb) {
        const bf16_t* aRow0 = Ab + (size_t)(rowBase + l)      * (size_t)ldab;
        const bf16_t* aRow1 = Ab + (size_t)(rowBase + 16 + l) * (size_t)ldab;
        const bf16_t* wRow0 = Wb + (size_t)(colBase + l)      * (size_t)ldwb;
        const bf16_t* wRow1 = Wb + (size_t)(colBase + 16 + l) * (size_t)ldwb;
        for (int k0 = 0; k0 < Kb; k0 += 32) {
            v16bf af0 = load_a_frag(aRow0 + k0, hi);
            v16bf af1 = load_a_frag(aRow1 + k0, hi);
            v16bf bf0 = load_b_frag(wRow0 + k0, hi);
            v16bf bf1 = load_b_frag(wRow1 + k0, hi);
            acc[0][0] = __builtin_amdgcn_wmma_f32_16x16x32_bf16(
                false, af0, false, bf0, (short)0, acc[0][0], false, false);
            acc[0][1] = __builtin_amdgcn_wmma_f32_16x16x32_bf16(
                false, af0, false, bf1, (short)0, acc[0][1], false, false);
            acc[1][0] = __builtin_amdgcn_wmma_f32_16x16x32_bf16(
                false, af1, false, bf0, (short)0, acc[1][0], false, false);
            acc[1][1] = __builtin_amdgcn_wmma_f32_16x16x32_bf16(
                false, af1, false, bf1, (short)0, acc[1][1], false, false);
        }
    };

    do_segment(A, lda, W, ldw, K);
    if (K2 > 0) do_segment(A2, lda2, W2, ldw2, K2);

#pragma unroll
    for (int mi = 0; mi < 2; ++mi) {
        const int row0 = rowBase + mi * 16 + hi * 8;
#pragma unroll
        for (int ni = 0; ni < 2; ++ni) {
            const int col = colBase + ni * 16 + l;
            const float bv = bias ? bias[col] : 0.0f;
            float* cp = C + (size_t)row0 * (size_t)ldc + col;
#pragma unroll
            for (int r = 0; r < 8; ++r) {
                float v = acc[mi][ni][r] + bv;
                if (accumulate) v += cp[(size_t)r * (size_t)ldc];
                cp[(size_t)r * (size_t)ldc] = v;
            }
        }
    }
}

// ---------------- LSTM cell kernels ----------------
// measure-level: g = xg[(b*M + t)] + rproj[b]; update c; h_bf = bf16(h);
// lat_bf[(b*M + t)] = bf16(tanh(h))
__global__ void measure_cell_kernel(const float* __restrict__ xg,
                                    const float* __restrict__ rp,
                                    float* __restrict__ c,
                                    bf16_t* __restrict__ h_bf,
                                    bf16_t* __restrict__ lat_bf,
                                    int t)
{
    int idx = blockIdx.x * blockDim.x + threadIdx.x;
    if (idx >= Bc * Hc) return;
    int b = idx / Hc, j = idx - b * Hc;
    const float* g4 = xg + (size_t)(b * Mm + t) * G4H;
    const float* r4 = rp + (size_t)b * G4H;
    float gi = g4[j]          + r4[j];
    float gf = g4[Hc + j]     + r4[Hc + j];
    float gg = g4[2 * Hc + j] + r4[2 * Hc + j];
    float go = g4[3 * Hc + j] + r4[3 * Hc + j];
    float cn = sigmoidf(gf) * c[idx] + sigmoidf(gi) * tanhf(gg);
    float hn = sigmoidf(go) * tanhf(cn);
    c[idx] = cn;
    h_bf[idx] = f2bf(hn);
    lat_bf[(size_t)(b * Mm + t) * Hc + j] = f2bf(tanhf(hn));
}

// beat-level: g = latproj[b] + rproj[b]; update c; h_bf for recurrence,
// th_bf = bf16(tanh(h)) for the output projection
__global__ void beat_cell_kernel(const float* __restrict__ lp,
                                 const float* __restrict__ rp,
                                 float* __restrict__ c,
                                 bf16_t* __restrict__ h_bf,
                                 bf16_t* __restrict__ th_bf)
{
    int idx = blockIdx.x * blockDim.x + threadIdx.x;
    if (idx >= Bc * Hc) return;
    int b = idx / Hc, j = idx - b * Hc;
    const float* l4 = lp + (size_t)b * G4H;
    const float* r4 = rp + (size_t)b * G4H;
    float gi = l4[j]          + r4[j];
    float gf = l4[Hc + j]     + r4[Hc + j];
    float gg = l4[2 * Hc + j] + r4[2 * Hc + j];
    float go = l4[3 * Hc + j] + r4[3 * Hc + j];
    float cn = sigmoidf(gf) * c[idx] + sigmoidf(gi) * tanhf(gg);
    float hn = sigmoidf(go) * tanhf(cn);
    c[idx] = cn;
    h_bf[idx]  = f2bf(hn);
    th_bf[idx] = f2bf(tanhf(hn));
}

// ---------------- host-side helpers ----------------
static inline void launch_gemm(hipStream_t s,
                               const bf16_t* A, int lda,
                               const bf16_t* W, int ldw, int K,
                               const bf16_t* A2, int lda2,
                               const bf16_t* W2, int ldw2, int K2,
                               float* C, int ldc, const float* bias,
                               int accumulate, int Mrows, int Ncols)
{
    dim3 grid(Ncols / 64, Mrows / 128);
    gemm_bf16_wmma<<<grid, 256, 0, s>>>(A, lda, W, ldw, K,
                                        A2, lda2, W2, ldw2, K2,
                                        C, ldc, bias, accumulate);
}

static inline void launch_cvt(hipStream_t s, const float* in, bf16_t* out, long n) {
    int blocks = (int)((n + 255) / 256); if (blocks > 4096) blocks = 4096;
    cvt_f32_bf16_kernel<<<blocks, 256, 0, s>>>(in, out, n);
}

extern "C" void kernel_launch(void* const* d_in, const int* in_sizes, int n_in,
                              void* d_out, int out_size, void* d_ws, size_t ws_size,
                              hipStream_t stream)
{
    const float* latent = (const float*)d_in[0];  // [B, M, IN]
    const float* inputs = (const float*)d_in[1];  // [B, M*S, O]
    const float* mWih   = (const float*)d_in[2];  // [4H, IN]
    const float* mWhh   = (const float*)d_in[3];  // [4H, H]
    const float* mb     = (const float*)d_in[4];  // [4H]
    const float* bWih   = (const float*)d_in[5];  // [4H, H+O]
    const float* bWhh   = (const float*)d_in[6];  // [4H, H]
    const float* bb     = (const float*)d_in[7];  // [4H]
    const float* linW   = (const float*)d_in[8];  // [O, H]
    const float* linb   = (const float*)d_in[9];  // [O]
    float* out = (float*)d_out;                   // [B, M*S, O]

    // ---- carve workspace (256B aligned blocks) ----
    char* p = (char*)d_ws;
    auto carve = [&](size_t elems, size_t esize) -> char* {
        char* r = p;
        p += (elems * esize + 255) & ~(size_t)255;
        return r;
    };
    bf16_t* latent_bf = (bf16_t*)carve((size_t)Bc * Mm * INc, 2);
    bf16_t* inputs_bf = (bf16_t*)carve((size_t)Bc * Tt * Oc, 2);
    bf16_t* mWih_bf   = (bf16_t*)carve((size_t)G4H * INc, 2);
    bf16_t* mWhh_bf   = (bf16_t*)carve((size_t)G4H * Hc, 2);
    bf16_t* bWih_bf   = (bf16_t*)carve((size_t)G4H * (Hc + Oc), 2);
    bf16_t* bWhh_bf   = (bf16_t*)carve((size_t)G4H * Hc, 2);
    bf16_t* linW_bf   = (bf16_t*)carve((size_t)Oc * Hc, 2);
    float*  xg_m      = (float*)carve((size_t)Bc * Mm * G4H, 4);   // 134 MB
    bf16_t* lat_bf    = (bf16_t*)carve((size_t)Bc * Mm * Hc, 2);
    float*  latproj   = (float*)carve((size_t)Bc * G4H, 4);
    float*  rproj     = (float*)carve((size_t)Bc * G4H, 4);
    bf16_t* tanh_h_bf = (bf16_t*)carve((size_t)Bc * Hc, 2);
    // contiguous state block (zero-filled every call)
    char* state0 = p;
    float*  c_m    = (float*)carve((size_t)Bc * Hc, 4);
    float*  c_b    = (float*)carve((size_t)Bc * Hc, 4);
    bf16_t* h_m_bf = (bf16_t*)carve((size_t)Bc * Hc, 2);
    bf16_t* h_b_bf = (bf16_t*)carve((size_t)Bc * Hc, 2);
    long state_dwords = (long)((p - state0) / 4);

    // ---- phase 0: conversions + state init ----
    launch_cvt(stream, latent, latent_bf, (long)Bc * Mm * INc);
    launch_cvt(stream, inputs, inputs_bf, (long)Bc * Tt * Oc);
    launch_cvt(stream, mWih, mWih_bf, (long)G4H * INc);
    launch_cvt(stream, mWhh, mWhh_bf, (long)G4H * Hc);
    launch_cvt(stream, bWih, bWih_bf, (long)G4H * (Hc + Oc));
    launch_cvt(stream, bWhh, bWhh_bf, (long)G4H * Hc);
    launch_cvt(stream, linW, linW_bf, (long)Oc * Hc);
    {
        int blocks = (int)((state_dwords + 255) / 256); if (blocks > 2048) blocks = 2048;
        fill_zero_kernel<<<blocks, 256, 0, stream>>>((uint32_t*)state0, state_dwords);
    }

    const int cellBlocks = (Bc * Hc + 255) / 256;

    // ---- phase 1: measure input projection (one big GEMM, +mb) ----
    // [B*M, IN] @ [4H, IN]^T -> [B*M, 4H]
    launch_gemm(stream, latent_bf, INc, mWih_bf, INc, INc,
                nullptr, 0, nullptr, 0, 0,
                xg_m, G4H, mb, 0, Bc * Mm, G4H);

    // ---- phase 2: measure LSTM (32 sequential steps) ----
    for (int t = 0; t < Mm; ++t) {
        // rproj = h @ mWhh^T   [256,1024] x [1024,4096]
        launch_gemm(stream, h_m_bf, Hc, mWhh_bf, Hc, Hc,
                    nullptr, 0, nullptr, 0, 0,
                    rproj, G4H, nullptr, 0, Bc, G4H);
        measure_cell_kernel<<<cellBlocks, 256, 0, stream>>>(
            xg_m, rproj, c_m, h_m_bf, lat_bf, t);
    }

    // ---- phase 3: beat LSTM (512 sequential steps) + output projection ----
    for (int t = 0; t < Tt; ++t) {
        const int m = t / Sc;
        if ((t % Sc) == 0) {
            // latproj = lat[:, m] @ bWih[:, :H]^T + bb  (reused for 16 steps)
            launch_gemm(stream, lat_bf + (size_t)m * Hc, Mm * Hc,
                        bWih_bf, Hc + Oc, Hc,
                        nullptr, 0, nullptr, 0, 0,
                        latproj, G4H, bb, 0, Bc, G4H);
        }
        // rproj = h @ bWhh^T + x_t @ bWih[:, H:]^T   (fused two-segment GEMM)
        launch_gemm(stream, h_b_bf, Hc, bWhh_bf, Hc, Hc,
                    inputs_bf + (size_t)t * Oc, Tt * Oc,
                    bWih_bf + Hc, Hc + Oc, Oc,
                    rproj, G4H, nullptr, 0, Bc, G4H);
        beat_cell_kernel<<<cellBlocks, 256, 0, stream>>>(
            latproj, rproj, c_b, h_b_bf, tanh_h_bf);
        // out[:, t] = tanh(h) @ linW^T + linb  [256,1024] x [1024,128]
        launch_gemm(stream, tanh_h_bf, Hc, linW_bf, Hc, Hc,
                    nullptr, 0, nullptr, 0, 0,
                    out + (size_t)t * Oc, Tt * Oc, linb, 0, Bc, Oc);
    }
}
